// AttentionBlock_40063454937384
// MI455X (gfx1250) — matrix operand order repaired and checked by reading, not compiled, not run on hardware
//
#include <hip/hip_runtime.h>

typedef __bf16 bf16;
typedef __attribute__((ext_vector_type(16))) __bf16 v16bf;
typedef __attribute__((ext_vector_type(8)))  __bf16 v8bf;
typedef __attribute__((ext_vector_type(8)))  float  v8f;

#define NB   8      // batch
#define CH   512    // channels
#define NH   8      // heads
#define DK   64     // head dim
#define NT   1024   // tokens = 32*32
#define QKVO 1536   // 3*NH*DK
#define SM_SCALE 0.125f   // 1/sqrt(64)

static __device__ __forceinline__ v16bf cat8(v8bf lo, v8bf hi) {
  return __builtin_shufflevector(lo, hi, 0,1,2,3,4,5,6,7,8,9,10,11,12,13,14,15);
}

// A operand (16x32 bf16). lane<16: row=lane, K={0..7,16..23}; lane>=16: row=lane-16, K={8..15,24..31}
static __device__ __forceinline__ v16bf load_A(const bf16* rowptr, int half) {
  v8bf lo = *reinterpret_cast<const v8bf*>(rowptr + half * 8);
  v8bf hi = *reinterpret_cast<const v8bf*>(rowptr + 16 + half * 8);
  return cat8(lo, hi);
}
// B operand (32x16 bf16). lane<16: col=lane, K={0..7,16..23}; lane>=16: col=lane-16, K={8..15,24..31}
static __device__ __forceinline__ v16bf load_B(const bf16* colptr, int half) {
  v8bf lo = *reinterpret_cast<const v8bf*>(colptr + half * 8);
  v8bf hi = *reinterpret_cast<const v8bf*>(colptr + 16 + half * 8);
  return cat8(lo, hi);
}

static __device__ __forceinline__ v8f wmma_bf16(v16bf a, v16bf b, v8f c) {
  return __builtin_amdgcn_wmma_f32_16x16x32_bf16(false, a, false, b, (short)0, c, false, false);
}

static __device__ __forceinline__ v8f splat8(float v) {
  v8f r;
#pragma unroll
  for (int i = 0; i < 8; ++i) r[i] = v;
  return r;
}

// ---------------- prep kernels ----------------

// x [B,C,N] f32 -> xfb [B,N,C] bf16 (c fastest -> coalesced writes)
__global__ void prep_xf(const float* __restrict__ x, bf16* __restrict__ xfb) {
  size_t i = (size_t)blockIdx.x * blockDim.x + threadIdx.x;
  if (i >= (size_t)NB * NT * CH) return;
  int c = (int)(i % CH);
  int n = (int)((i / CH) % NT);
  int b = (int)(i / ((size_t)CH * NT));
  xfb[i] = (bf16)x[((size_t)b * CH + c) * NT + n];
}

// w [rows=in, cols=out] f32 -> wt [out, in] bf16
__global__ void prep_wt(const float* __restrict__ w, bf16* __restrict__ wt, int rows, int cols) {
  size_t i = (size_t)blockIdx.x * blockDim.x + threadIdx.x;
  if (i >= (size_t)rows * cols) return;
  int r = (int)(i % rows);   // in-dim (fastest in wt)
  int c = (int)(i / rows);   // out-dim
  wt[i] = (bf16)w[(size_t)r * cols + c];
}

// ---------------- QKV projection (2x2 register-blocked) ----------------
// qkv = xf @ Wqkv + b ; scatter to q[B,h,N,64], k[B,h,N,64], vT[B,h,64,N] in bf16
static __device__ __forceinline__ void qkv_scatter(
    v8f acc, int b, int tokbase, int half, int col,
    bf16* __restrict__ qb, bf16* __restrict__ kb, bf16* __restrict__ vtb) {
  int head = col / 192;
  int t = col - head * 192;    // 0..63 -> q, 64..127 -> k, 128..191 -> v
#pragma unroll
  for (int r = 0; r < 8; ++r) {
    int tok = tokbase + r + half * 8;
    bf16 v = (bf16)acc[r];
    if (t < 64)       qb[(((size_t)b * NH + head) * NT + tok) * DK + t] = v;
    else if (t < 128) kb[(((size_t)b * NH + head) * NT + tok) * DK + (t - 64)] = v;
    else              vtb[(((size_t)b * NH + head) * DK + (t - 128)) * NT + tok] = v;
  }
}

__global__ void __launch_bounds__(256) qkv_gemm(
    const bf16* __restrict__ xfb, const bf16* __restrict__ wt,
    const float* __restrict__ bqkv,
    bf16* __restrict__ qb, bf16* __restrict__ kb, bf16* __restrict__ vtb) {
  int wave = (int)((blockIdx.x * blockDim.x + threadIdx.x) >> 5);
  int lane = threadIdx.x & 31;
  int l16 = lane & 15, half = lane >> 4;
  int ct2 = wave % (QKVO / 32);               // 32-col block (0..47)
  int nt2 = (wave / (QKVO / 32)) % (NT / 32); // 32-token block (0..31)
  int b   = wave / ((QKVO / 32) * (NT / 32));
  if (b >= NB) return;

  int row0 = nt2 * 32 + l16;        // A lanes, tile 0
  int col0 = ct2 * 32 + l16;        // B lanes, tile 0
  const bf16* arow0 = xfb + ((size_t)b * NT + row0) * CH;
  const bf16* arow1 = arow0 + (size_t)16 * CH;
  const bf16* brow0 = wt + (size_t)col0 * CH;
  const bf16* brow1 = brow0 + (size_t)16 * CH;

  v8f acc00 = splat8(bqkv[col0]),      acc01 = splat8(bqkv[col0 + 16]);
  v8f acc10 = acc00,                   acc11 = acc01;
#pragma unroll
  for (int kc = 0; kc < CH / 32; ++kc) {
    v16bf a0 = load_A(arow0 + kc * 32, half);
    v16bf a1 = load_A(arow1 + kc * 32, half);
    v16bf b0 = load_B(brow0 + kc * 32, half);
    v16bf b1 = load_B(brow1 + kc * 32, half);
    acc00 = wmma_bf16(a0, b0, acc00);
    acc01 = wmma_bf16(a0, b1, acc01);
    acc10 = wmma_bf16(a1, b0, acc10);
    acc11 = wmma_bf16(a1, b1, acc11);
  }

  qkv_scatter(acc00, b, nt2 * 32,      half, col0,      qb, kb, vtb);
  qkv_scatter(acc01, b, nt2 * 32,      half, col0 + 16, qb, kb, vtb);
  qkv_scatter(acc10, b, nt2 * 32 + 16, half, col0,      qb, kb, vtb);
  qkv_scatter(acc11, b, nt2 * 32 + 16, half, col0 + 16, qb, kb, vtb);
}

// ---------------- flash attention ----------------
// one wave per (b, h, 16-query tile); online softmax over 1024 keys in 32-key tiles
__global__ void __launch_bounds__(128) attn_kernel(
    const bf16* __restrict__ qb, const bf16* __restrict__ kb,
    const bf16* __restrict__ vtb, bf16* __restrict__ resO) {
  __shared__ bf16 pstage[4][16 * 32];
  int wslot = threadIdx.x >> 5;
  int wave  = blockIdx.x * 4 + wslot;
  int lane  = threadIdx.x & 31;
  int l16 = lane & 15, half = lane >> 4;
  int qt = wave % (NT / 16);
  int h  = (wave / (NT / 16)) % NH;
  int b  = wave / ((NT / 16) * NH);
  if (b >= NB) return;

  const bf16* qbase = qb  + ((size_t)b * NH + h) * NT * DK;
  const bf16* kbase = kb  + ((size_t)b * NH + h) * NT * DK;
  const bf16* vbase = vtb + ((size_t)b * NH + h) * DK * NT;

  int qrow = qt * 16 + l16;
  v16bf qa0 = load_A(qbase + (size_t)qrow * DK, half);       // d 0..31
  v16bf qa1 = load_A(qbase + (size_t)qrow * DK + 32, half);  // d 32..63

  float m[8], l[8];
  v8f o0 = splat8(0.f), o1 = splat8(0.f), o2 = splat8(0.f), o3 = splat8(0.f);
#pragma unroll
  for (int r = 0; r < 8; ++r) { m[r] = -1e30f; l[r] = 0.f; }

  bf16* pl = pstage[wslot];

  for (int j0 = 0; j0 < NT; j0 += 32) {
    // prefetch next key tile (K rows + V rows) into cache: global_prefetch path
    if (j0 + 32 < NT) {
      __builtin_prefetch(kbase + (size_t)(j0 + 32 + l16) * DK, 0, 3);
      __builtin_prefetch(kbase + (size_t)(j0 + 48 + l16) * DK, 0, 3);
      __builtin_prefetch(vbase + (size_t)(l16 + half * 16) * NT + j0 + 32, 0, 3);
      __builtin_prefetch(vbase + (size_t)(l16 + half * 16 + 32) * NT + j0 + 32, 0, 3);
    }
    // scores S = Q Kt for 32 keys (two 16-col tiles), K-dim = 64 in two chunks
    v8f s0 = splat8(0.f), s1 = splat8(0.f);
    {
      const bf16* kr0 = kbase + (size_t)(j0 + l16) * DK;
      s0 = wmma_bf16(qa0, load_B(kr0, half), s0);
      s0 = wmma_bf16(qa1, load_B(kr0 + 32, half), s0);
      const bf16* kr1 = kbase + (size_t)(j0 + 16 + l16) * DK;
      s1 = wmma_bf16(qa0, load_B(kr1, half), s1);
      s1 = wmma_bf16(qa1, load_B(kr1 + 32, half), s1);
    }
    // scale + row max over this tile (reduce across the 16-lane halves)
    float mt[8];
#pragma unroll
    for (int r = 0; r < 8; ++r) {
      s0[r] *= SM_SCALE; s1[r] *= SM_SCALE;
      mt[r] = fmaxf(s0[r], s1[r]);
    }
#pragma unroll
    for (int mk = 1; mk < 16; mk <<= 1) {
#pragma unroll
      for (int r = 0; r < 8; ++r) mt[r] = fmaxf(mt[r], __shfl_xor(mt[r], mk, 32));
    }
    float alpha[8];
#pragma unroll
    for (int r = 0; r < 8; ++r) {
      float mn = fmaxf(m[r], mt[r]);
      alpha[r] = __expf(m[r] - mn);
      m[r] = mn;
      s0[r] = __expf(s0[r] - mn);   // now P
      s1[r] = __expf(s1[r] - mn);
    }
    float rs[8];
#pragma unroll
    for (int r = 0; r < 8; ++r) rs[r] = s0[r] + s1[r];
#pragma unroll
    for (int mk = 1; mk < 16; mk <<= 1) {
#pragma unroll
      for (int r = 0; r < 8; ++r) rs[r] += __shfl_xor(rs[r], mk, 32);
    }
#pragma unroll
    for (int r = 0; r < 8; ++r) {
      l[r] = l[r] * alpha[r] + rs[r];
      o0[r] *= alpha[r]; o1[r] *= alpha[r]; o2[r] *= alpha[r]; o3[r] *= alpha[r];
    }
    // stage P (C-layout f32) -> LDS bf16 row-major [16][32], reload in A layout
#pragma unroll
    for (int r = 0; r < 8; ++r) {
      int prow = r + half * 8;
      pl[prow * 32 + l16]      = (bf16)s0[r];
      pl[prow * 32 + 16 + l16] = (bf16)s1[r];
    }
    v16bf pa = load_A(pl + l16 * 32, half);
    // O += P @ V  (vT rows are d-indices, contiguous along keys)
    o0 = wmma_bf16(pa, load_B(vbase + (size_t)(0 * 16 + l16) * NT + j0, half), o0);
    o1 = wmma_bf16(pa, load_B(vbase + (size_t)(1 * 16 + l16) * NT + j0, half), o1);
    o2 = wmma_bf16(pa, load_B(vbase + (size_t)(2 * 16 + l16) * NT + j0, half), o2);
    o3 = wmma_bf16(pa, load_B(vbase + (size_t)(3 * 16 + l16) * NT + j0, half), o3);
  }

#pragma unroll
  for (int r = 0; r < 8; ++r) {
    float inv = 1.0f / l[r];
    o0[r] *= inv; o1[r] *= inv; o2[r] *= inv; o3[r] *= inv;
  }
  // resO [B, N, 512] bf16, cols h*64 + dchunk*16 + l16
#pragma unroll
  for (int r = 0; r < 8; ++r) {
    int tok = qt * 16 + r + half * 8;
    bf16* dst = resO + ((size_t)b * NT + tok) * CH + h * DK;
    dst[0 * 16 + l16] = (bf16)o0[r];
    dst[1 * 16 + l16] = (bf16)o1[r];
    dst[2 * 16 + l16] = (bf16)o2[r];
    dst[3 * 16 + l16] = (bf16)o3[r];
  }
}

// ---------------- output projection + residual (2x2 register-blocked) ----------------
__global__ void __launch_bounds__(256) out_proj(
    const bf16* __restrict__ resO, const bf16* __restrict__ woutT,
    const float* __restrict__ bout, const float* __restrict__ x,
    float* __restrict__ out) {
  int wave = (int)((blockIdx.x * blockDim.x + threadIdx.x) >> 5);
  int lane = threadIdx.x & 31;
  int l16 = lane & 15, half = lane >> 4;
  int ct2 = wave % (CH / 32);
  int nt2 = (wave / (CH / 32)) % (NT / 32);
  int b   = wave / ((CH / 32) * (NT / 32));
  if (b >= NB) return;

  int row0 = nt2 * 32 + l16;   // token, tile 0
  int col0 = ct2 * 32 + l16;   // output channel, tile 0
  const bf16* arow0 = resO + ((size_t)b * NT + row0) * CH;
  const bf16* arow1 = arow0 + (size_t)16 * CH;
  const bf16* brow0 = woutT + (size_t)col0 * CH;
  const bf16* brow1 = brow0 + (size_t)16 * CH;

  v8f acc00 = splat8(bout[col0]), acc01 = splat8(bout[col0 + 16]);
  v8f acc10 = acc00,              acc11 = acc01;
#pragma unroll
  for (int kc = 0; kc < CH / 32; ++kc) {
    v16bf a0 = load_A(arow0 + kc * 32, half);
    v16bf a1 = load_A(arow1 + kc * 32, half);
    v16bf b0 = load_B(brow0 + kc * 32, half);
    v16bf b1 = load_B(brow1 + kc * 32, half);
    acc00 = wmma_bf16(a0, b0, acc00);
    acc01 = wmma_bf16(a0, b1, acc01);
    acc10 = wmma_bf16(a1, b0, acc10);
    acc11 = wmma_bf16(a1, b1, acc11);
  }

  // out[b, col, tok] = acc + x[b, col, tok]  (residual in original layout)
#pragma unroll
  for (int r = 0; r < 8; ++r) {
    int tok0 = nt2 * 32 + r + half * 8;
    int tok1 = tok0 + 16;
    size_t i00 = ((size_t)b * CH + col0) * NT + tok0;
    size_t i01 = ((size_t)b * CH + col0 + 16) * NT + tok0;
    size_t i10 = ((size_t)b * CH + col0) * NT + tok1;
    size_t i11 = ((size_t)b * CH + col0 + 16) * NT + tok1;
    out[i00] = acc00[r] + x[i00];
    out[i01] = acc01[r] + x[i01];
    out[i10] = acc10[r] + x[i10];
    out[i11] = acc11[r] + x[i11];
  }
}

// ---------------- launch ----------------
extern "C" void kernel_launch(void* const* d_in, const int* in_sizes, int n_in,
                              void* d_out, int out_size, void* d_ws, size_t ws_size,
                              hipStream_t stream) {
  const float* x    = (const float*)d_in[0];
  const float* Wqkv = (const float*)d_in[1];
  const float* bqkv = (const float*)d_in[2];
  const float* Wout = (const float*)d_in[3];
  const float* bout = (const float*)d_in[4];
  float* out = (float*)d_out;

  char* ws = (char*)d_ws;
  bf16* xfb   = (bf16*)(ws + 0);          //  8 MB : xf bf16 [B,N,C]
  bf16* wqkvt = (bf16*)(ws + 8388608);    // 1.5 MB: Wqkv^T bf16 [1536,512]
  bf16* woutt = (bf16*)(ws + 9961472);    // 0.5 MB: Wout^T bf16 [512,512]
  bf16* qb    = (bf16*)(ws + 10485760);   //  8 MB : q [B,h,N,64]
  bf16* kb    = (bf16*)(ws + 18874368);   //  8 MB : k [B,h,N,64]
  bf16* vtb   = (bf16*)(ws + 27262976);   //  8 MB : v^T [B,h,64,N]
  bf16* resO  = (bf16*)(ws + 35651584);   //  8 MB : attn out [B,N,512]

  {
    size_t n = (size_t)NB * NT * CH;
    prep_xf<<<(unsigned)((n + 255) / 256), 256, 0, stream>>>(x, xfb);
  }
  prep_wt<<<(CH * QKVO + 255) / 256, 256, 0, stream>>>(Wqkv, wqkvt, CH, QKVO);
  prep_wt<<<(CH * CH + 255) / 256, 256, 0, stream>>>(Wout, woutt, CH, CH);

  // 8 waves / 256-thread block; each wave computes a 32x32 output block
  qkv_gemm<<<(NB * (NT / 32) * (QKVO / 32)) / 8, 256, 0, stream>>>(xfb, wqkvt, bqkv, qb, kb, vtb);
  // 4 waves / 128-thread block
  attn_kernel<<<(NB * NH * (NT / 16)) / 4, 128, 0, stream>>>(qb, kb, vtb, resO);
  out_proj<<<(NB * (NT / 32) * (CH / 32)) / 8, 256, 0, stream>>>(resO, woutt, bout, x, out);
}
